// LogAnomaly_9259949490921
// MI455X (gfx1250) — matrix operand order repaired and checked by reading, not compiled, not run on hardware
//
#include <hip/hip_runtime.h>
#include <hip/hip_bf16.h>
#include <stdint.h>

// Problem sizes (fixed by the reference).
#define NB 2048   // windows (batch)
#define NT 64     // time steps
#define NH 128    // hidden
#define NG 512    // 4*NH gates
#define NE 1024   // events / classifier outputs
#define BM 16     // batch rows per block (one WMMA M tile)

typedef __attribute__((ext_vector_type(16))) __bf16 v16bf;
typedef __attribute__((ext_vector_type(8)))  float  v8f;

__device__ __forceinline__ uint16_t f2bf(float f) {
  uint32_t u = __float_as_uint(f);
  u += 0x7FFFu + ((u >> 16) & 1u);   // round-to-nearest-even
  return (uint16_t)(u >> 16);
}

__device__ __forceinline__ float sigf(float x) {
  return 1.0f / (1.0f + __expf(-x));
}

// ---- WMMA operand loaders (bf16 16x16x32) -------------------------------
// A operand (16x32, MxK), ISA 7.12.2 16-bit A layout:
//   lane l -> row m = l&15 ; VGPR v holds K pair base (v&3)*2 + (v>>2)*16 + (l>>4)*8
__device__ __forceinline__ v16bf load_opA(const uint16_t* base, int rs, int row,
                                          int kb, int lane) {
  union { v16bf v; uint32_t u[8]; } r;
  const uint16_t* p = base + row * rs + kb + (((lane >> 4) & 1) << 3);
#pragma unroll
  for (int v = 0; v < 8; ++v) {
    int k = ((v & 3) << 1) + ((v >> 2) << 4);
    r.u[v] = *reinterpret_cast<const uint32_t*>(p + k);
  }
  return r.v;
}

// B operand (32x16, KxN) for gates = x @ W^T: one B column == one row of the
// row-major weight matrix. Per the ISA sparse-B layout analogy:
//   lane l -> col n = l&15 ; VGPR v holds K pair base (v&3)*2 + (v>>2)*16 + (l>>4)*8
// -> same K order as load_opA.
__device__ __forceinline__ v16bf load_opB(const uint16_t* base, int rs, int row,
                                          int kb, int lane) {
  union { v16bf v; uint32_t u[8]; } r;
  const uint16_t* p = base + row * rs + kb + (((lane >> 4) & 1) << 3);
#pragma unroll
  for (int v = 0; v < 8; ++v)
    r.u[v] = *reinterpret_cast<const uint32_t*>(p + ((v & 3) << 1) + ((v >> 2) << 4));
  return r.v;
}

// ---- small helper kernels ----------------------------------------------
__global__ __launch_bounds__(256) void f2bf_kernel(const float* __restrict__ in,
                                                   uint16_t* __restrict__ out, int n) {
  int i = blockIdx.x * 256 + threadIdx.x;
  if (i < n) out[i] = f2bf(in[i]);
}

// quant_Wih0 [512][1024] -> [1024][512] so a histogram column gather is contiguous.
__global__ __launch_bounds__(256) void transpose_wih_kernel(const float* __restrict__ in,
                                                            float* __restrict__ out) {
  int i = blockIdx.x * 256 + threadIdx.x;
  if (i < NG * NE) { int r = i / NE; int e = i - r * NE; out[e * NG + r] = in[i]; }
}

// ---- persistent LSTM layer kernel --------------------------------------
// MODE 0: scalar input  (seq layer 0)   px = e_t * Wih[:,0]
// MODE 1: histogram input (quant layer 0) px accumulated via WihT column gather
// MODE 2: dense input (layer 1)          px via second WMMA chain on prev-layer h
template <int MODE>
__global__ __launch_bounds__(256) void lstm_kernel(
    const int* seq,            // [NB][NT]      (MODE 0/1)
    const uint16_t* xin,       // [NB][NT][NH]  bf16 (MODE 2)
    const float* wih_scalar,   // [NG]          (MODE 0)
    const float* wihT,         // [NE][NG] f32  (MODE 1)
    const uint16_t* wihB,      // [NG][NH] bf16 (MODE 2)
    const uint16_t* whhB,      // [NG][NH] bf16
    const float* bih, const float* bhh,
    uint16_t* hout,            // [NB][NT][NH] bf16 (may alias xin; disjoint in time)
    uint16_t* hlast)           // [NB][NH] bf16 or nullptr
{
  __shared__ uint16_t hl[BM * NH];                       // current h, bf16
  __shared__ float    bias_l[NG];
  __shared__ uint16_t xb[(MODE == 2) ? BM * NH : 2];     // staged prev-layer h
  __shared__ float    pxacc[(MODE == 1) ? BM * NG : 1];  // running input proj
  __shared__ float    w0[(MODE == 0) ? NG : 1];
  __shared__ float    eb[BM];
  __shared__ int      ei[BM];

  const int tid  = threadIdx.x;
  const int lane = tid & 31;
  const int w    = tid >> 5;              // wave id, 0..7
  const int b0   = blockIdx.x * BM;

  for (int i = tid; i < NG; i += 256) bias_l[i] = bih[i] + bhh[i];
  if (MODE == 0) for (int i = tid; i < NG; i += 256) w0[i] = wih_scalar[i];
  if (MODE == 1) for (int i = tid; i < BM * NG; i += 256) pxacc[i] = 0.0f;
  for (int i = tid; i < BM * NH; i += 256) hl[i] = 0;    // h0 = 0

  float creg[8];
#pragma unroll
  for (int r = 0; r < 8; ++r) creg[r] = 0.0f;            // c0 = 0

  const int n15 = lane & 15;
  const int mhi = (lane >> 4) << 3;       // row offset 0 or 8 in C/D layout
  const int j   = (w << 4) + n15;         // hidden unit this lane owns

  for (int t = 0; t < NT; ++t) {
    // (a) stage per-step inputs
    if (MODE == 0) { if (tid < BM) eb[tid] = (float)seq[(b0 + tid) * NT + t]; }
    if (MODE == 1) { if (tid < BM) ei[tid] = seq[(b0 + tid) * NT + t]; }
    if (MODE == 2) {
      const uint32_t* src = reinterpret_cast<const uint32_t*>(xin);
      uint32_t* dst = reinterpret_cast<uint32_t*>(xb);
      for (int i = tid; i < BM * NH / 2; i += 256) {
        int m = i >> 6, q = i & 63;
        dst[i] = src[((size_t)(b0 + m) * NT + t) * (NH / 2) + q];
      }
    }
    __syncthreads();

    // (b) histogram: incremental input projection (replaces [*,1024]x[1024,512] GEMM)
    if (MODE == 1) {
#pragma unroll 1
      for (int m = 0; m < BM; ++m) {
        const float* col = wihT + (size_t)ei[m] * NG;
        for (int i = tid; i < NG; i += 256) pxacc[m * NG + i] += col[i];
      }
      __syncthreads();
    }

    // (c) gate GEMM: each wave owns tiles {w, 8+w, 16+w, 24+w} -> i,f,g,o cols j
    float gacc[4][8];
#pragma unroll
    for (int gb = 0; gb < 4; ++gb) {
      const int tile = gb * 8 + w;
      const int colg = tile * 16 + n15;     // == gb*128 + j
      v8f acc;
      const float bv = bias_l[colg];
#pragma unroll
      for (int r = 0; r < 8; ++r) acc[r] = bv;
#pragma unroll
      for (int kk = 0; kk < NH; kk += 32) {
        v16bf a = load_opA(hl, NH, n15, kk, lane);
        v16bf b = load_opB(whhB + (size_t)tile * 16 * NH, NH, n15, kk, lane);
        acc = __builtin_amdgcn_wmma_f32_16x16x32_bf16(false, a, false, b,
                                                      (short)0, acc, false, false);
      }
      if (MODE == 2) {
#pragma unroll
        for (int kk = 0; kk < NH; kk += 32) {
          v16bf a = load_opA(xb, NH, n15, kk, lane);
          v16bf b = load_opB(wihB + (size_t)tile * 16 * NH, NH, n15, kk, lane);
          acc = __builtin_amdgcn_wmma_f32_16x16x32_bf16(false, a, false, b,
                                                        (short)0, acc, false, false);
        }
      }
#pragma unroll
      for (int r = 0; r < 8; ++r) {
        const int m = r + mhi;
        float v = acc[r];
        if (MODE == 0) v += eb[m] * w0[colg];
        if (MODE == 1) v += pxacc[m * NG + colg];
        gacc[gb][r] = v;
      }
    }
    __syncthreads();   // all waves done reading hl/xb before h is overwritten

    // (d) LSTM elementwise: (i,f,g,o) for unit j, rows m -- all in registers
#pragma unroll
    for (int r = 0; r < 8; ++r) {
      const int m = r + mhi;
      float iv = sigf(gacc[0][r]);
      float fv = sigf(gacc[1][r]);
      float gv = tanhf(gacc[2][r]);
      float ov = sigf(gacc[3][r]);
      float c  = fv * creg[r] + iv * gv;
      creg[r]  = c;
      float h  = ov * tanhf(c);
      uint16_t hb = f2bf(h);
      hl[m * NH + j] = hb;
      hout[((size_t)(b0 + m) * NT + t) * NH + j] = hb;
      if (hlast && t == NT - 1) hlast[(size_t)(b0 + m) * NH + j] = hb;
    }
    // next iteration's __syncthreads() publishes hl before it is read again
  }
}

// ---- classifier: out = [h_seq | h_quant] @ cls_W^T + cls_b --------------
__global__ __launch_bounds__(256) void cls_kernel(
    const uint16_t* __restrict__ hs, const uint16_t* __restrict__ hq,
    const uint16_t* __restrict__ wb,   // cls_W bf16 [NE][2*NH]
    const float* __restrict__ bias, float* __restrict__ out)
{
  __shared__ uint16_t al[BM * 2 * NH];   // 16 x 256 concat activations
  const int tid = threadIdx.x, lane = tid & 31, w = tid >> 5;
  const int b0 = blockIdx.x * BM, n0 = blockIdx.y * 128;

  uint32_t* ad = reinterpret_cast<uint32_t*>(al);
  const uint32_t* ss = reinterpret_cast<const uint32_t*>(hs);
  const uint32_t* sq = reinterpret_cast<const uint32_t*>(hq);
  for (int i = tid; i < BM * NH / 2; i += 256) {
    int m = i >> 6, q = i & 63;
    ad[m * 128 + q]      = ss[(b0 + m) * 64 + q];
    ad[m * 128 + 64 + q] = sq[(b0 + m) * 64 + q];
  }
  __syncthreads();

  const int n15 = lane & 15, mhi = (lane >> 4) << 3;
  const int colg = n0 + (w << 4) + n15;
  v8f acc;
  const float bv = bias[colg];
#pragma unroll
  for (int r = 0; r < 8; ++r) acc[r] = bv;
#pragma unroll
  for (int kk = 0; kk < 2 * NH; kk += 32) {
    v16bf a = load_opA(al, 2 * NH, n15, kk, lane);
    v16bf b = load_opB(wb + (size_t)(n0 + (w << 4)) * (2 * NH), 2 * NH, n15, kk, lane);
    acc = __builtin_amdgcn_wmma_f32_16x16x32_bf16(false, a, false, b,
                                                  (short)0, acc, false, false);
  }
#pragma unroll
  for (int r = 0; r < 8; ++r)
    out[(size_t)(b0 + r + mhi) * NE + colg] = acc[r];
}

// ---- launch ------------------------------------------------------------
extern "C" void kernel_launch(void* const* d_in, const int* in_sizes, int n_in,
                              void* d_out, int out_size, void* d_ws, size_t ws_size,
                              hipStream_t stream) {
  (void)in_sizes; (void)n_in; (void)out_size; (void)ws_size;

  const int*   seq   = (const int*)  d_in[0];
  const float* sWih0 = (const float*)d_in[1];
  const float* sWhh0 = (const float*)d_in[2];
  const float* sbih0 = (const float*)d_in[3];
  const float* sbhh0 = (const float*)d_in[4];
  const float* sWih1 = (const float*)d_in[5];
  const float* sWhh1 = (const float*)d_in[6];
  const float* sbih1 = (const float*)d_in[7];
  const float* sbhh1 = (const float*)d_in[8];
  const float* qWih0 = (const float*)d_in[9];
  const float* qWhh0 = (const float*)d_in[10];
  const float* qbih0 = (const float*)d_in[11];
  const float* qbhh0 = (const float*)d_in[12];
  const float* qWih1 = (const float*)d_in[13];
  const float* qWhh1 = (const float*)d_in[14];
  const float* qbih1 = (const float*)d_in[15];
  const float* qbhh1 = (const float*)d_in[16];
  const float* clsW  = (const float*)d_in[17];
  const float* clsb  = (const float*)d_in[18];
  float* out = (float*)d_out;

  // Workspace layout (~38 MB total).
  char* ws = (char*)d_ws;
  size_t off = 0;
  float*    wihT  = (float*)   (ws + off); off += (size_t)NE * NG * 4;       // 2 MB
  uint16_t* hbuf  = (uint16_t*)(ws + off); off += (size_t)NB * NT * NH * 2;  // 32 MB
  uint16_t* hls   = (uint16_t*)(ws + off); off += (size_t)NB * NH * 2;
  uint16_t* hlq   = (uint16_t*)(ws + off); off += (size_t)NB * NH * 2;
  uint16_t* whhS0 = (uint16_t*)(ws + off); off += (size_t)NG * NH * 2;
  uint16_t* whhS1 = (uint16_t*)(ws + off); off += (size_t)NG * NH * 2;
  uint16_t* whhQ0 = (uint16_t*)(ws + off); off += (size_t)NG * NH * 2;
  uint16_t* whhQ1 = (uint16_t*)(ws + off); off += (size_t)NG * NH * 2;
  uint16_t* wihS1 = (uint16_t*)(ws + off); off += (size_t)NG * NH * 2;
  uint16_t* wihQ1 = (uint16_t*)(ws + off); off += (size_t)NG * NH * 2;
  uint16_t* clsWb = (uint16_t*)(ws + off); off += (size_t)NE * 2 * NH * 2;

  // One-time weight prep (bf16 conversion + histogram-weight transpose).
  auto cv = [&](const float* s, uint16_t* d, int n) {
    hipLaunchKernelGGL(f2bf_kernel, dim3((n + 255) / 256), dim3(256), 0, stream, s, d, n);
  };
  cv(sWhh0, whhS0, NG * NH);  cv(sWhh1, whhS1, NG * NH);
  cv(qWhh0, whhQ0, NG * NH);  cv(qWhh1, whhQ1, NG * NH);
  cv(sWih1, wihS1, NG * NH);  cv(qWih1, wihQ1, NG * NH);
  cv(clsW,  clsWb, NE * 2 * NH);
  hipLaunchKernelGGL(transpose_wih_kernel, dim3((NG * NE + 255) / 256), dim3(256),
                     0, stream, qWih0, wihT);

  const dim3 lgrid(NB / BM), lblk(256);
  // seq branch
  hipLaunchKernelGGL((lstm_kernel<0>), lgrid, lblk, 0, stream,
                     seq, (const uint16_t*)nullptr, sWih0, (const float*)nullptr,
                     (const uint16_t*)nullptr, whhS0, sbih0, sbhh0, hbuf,
                     (uint16_t*)nullptr);
  hipLaunchKernelGGL((lstm_kernel<2>), lgrid, lblk, 0, stream,
                     (const int*)nullptr, hbuf, (const float*)nullptr,
                     (const float*)nullptr, wihS1, whhS1, sbih1, sbhh1, hbuf, hls);
  // quant branch (reuses hbuf)
  hipLaunchKernelGGL((lstm_kernel<1>), lgrid, lblk, 0, stream,
                     seq, (const uint16_t*)nullptr, (const float*)nullptr, wihT,
                     (const uint16_t*)nullptr, whhQ0, qbih0, qbhh0, hbuf,
                     (uint16_t*)nullptr);
  hipLaunchKernelGGL((lstm_kernel<2>), lgrid, lblk, 0, stream,
                     (const int*)nullptr, hbuf, (const float*)nullptr,
                     (const float*)nullptr, wihQ1, whhQ1, qbih1, qbhh1, hbuf, hlq);
  // classifier
  hipLaunchKernelGGL(cls_kernel, dim3(NB / BM, NE / 128), dim3(256), 0, stream,
                     hls, hlq, clsWb, clsb, out);
}
